// FeCAM_Manager_30262339568163
// MI455X (gfx1250) — matrix operand order repaired and checked by reading, not compiled, not run on hardware
//
#include <hip/hip_runtime.h>
#include <hip/hip_bf16.h>

typedef __attribute__((ext_vector_type(16))) __bf16 v16bf;
typedef __attribute__((ext_vector_type(8)))  __bf16 v8bf;
typedef __attribute__((ext_vector_type(8)))  float  v8f;

#define Bn 1024
#define Cn 100
#define Dn 768
#define BLK_ROWS 128
#define TPB 256
#define NHALF 384            // N range per wave (768/2)

union FragBF { v16bf v; v8bf h[2]; };

// ---------------------------------------------------------------------------
// Kernel 1: convert + transpose class_inv_covs [C,D,D] f32 -> invT[C,e,d] bf16
// ---------------------------------------------------------------------------
__global__ __launch_bounds__(256) void fecam_cvt_transpose(
    const float* __restrict__ inv, __bf16* __restrict__ invT)
{
  __shared__ float tile[32][33];
  const int tilesPerDim = Dn / 32;          // 24
  int bid = blockIdx.x;
  int c  = bid / (tilesPerDim * tilesPerDim);
  int t2 = bid % (tilesPerDim * tilesPerDim);
  int d0 = (t2 / tilesPerDim) * 32;
  int e0 = (t2 % tilesPerDim) * 32;
  int tx = threadIdx.x & 31;
  int ty = threadIdx.x >> 5;
  const float* src = inv + (size_t)c * Dn * Dn;
  #pragma unroll
  for (int r = 0; r < 32; r += 8)
    tile[ty + r][tx] = src[(size_t)(d0 + ty + r) * Dn + e0 + tx];
  __syncthreads();
  __bf16* dst = invT + (size_t)c * Dn * Dn;
  #pragma unroll
  for (int r = 0; r < 32; r += 8)
    dst[(size_t)(e0 + ty + r) * Dn + d0 + tx] = (__bf16)tile[tx][ty + r];
}

// ---------------------------------------------------------------------------
// Kernel 2: per (class, 128-row block). 8 waves = 4 M-strips (32 rows) x
// 2 N-halves. 2x2 register tiling: 4 WMMAs reuse 2 A-frags and 2 B-frags
// -> 1 global_b128 + 1 ds_b128 per WMMA. Quadratic-form fold fused per tile;
// N-half partials merged via ds_add_f32.
// ---------------------------------------------------------------------------
__global__ __launch_bounds__(256) void fecam_score_wmma(
    const float* __restrict__ raw, const float* __restrict__ means,
    const float* __restrict__ diags, const __bf16* __restrict__ invT,
    float* __restrict__ out)
{
  extern __shared__ char smem[];
  __bf16* A    = (__bf16*)smem;                               // [BLK_ROWS][Dn]
  float*  sInv = (float*)(smem + (size_t)BLK_ROWS * Dn * 2);  // [Dn]
  float*  m2   = sInv + Dn;                                   // [Dn]
  float*  scorebuf = m2 + Dn;                                 // [BLK_ROWS]

  const int bid = blockIdx.x;
  const int cls = bid / (Bn / BLK_ROWS);
  const int rb  = bid % (Bn / BLK_ROWS);
  const int t   = threadIdx.x;

  // Phase A: per-class scale vectors
  for (int d = t; d < Dn; d += TPB) {
    float s = 1.0f / diags[(size_t)cls * Dn + d];
    sInv[d] = s;
    m2[d]   = means[(size_t)cls * Dn + d] * s;
  }
  if (t < BLK_ROWS) scorebuf[t] = 0.0f;
  __syncthreads();

  // Phase B: bf16 diff block (128 x 768) into LDS
  const float* rawBlk = raw + (size_t)(rb * BLK_ROWS) * Dn;
  for (int idx = t; idx < BLK_ROWS * Dn; idx += TPB) {
    int d = idx % Dn;
    A[idx] = (__bf16)(rawBlk[idx] * sInv[d] - m2[d]);
  }
  __syncthreads();

  // Phase C: 2x2-tiled WMMA GEMM + fused reduction
  const int wave  = t >> 5;
  const int lane  = t & 31;
  const int strip = wave & 3;               // M strip (32 rows each)
  const int nhalf = wave >> 2;              // N half (384 wide)
  const int nr    = lane & 15;
  const int hs    = lane >> 4;
  const int rowBase = strip * 32;

  const __bf16* Arow0 = A + (size_t)(rowBase + nr) * Dn + hs * 16;
  const __bf16* Arow1 = Arow0 + (size_t)16 * Dn;
  const __bf16* Binv  = invT + (size_t)cls * Dn * Dn;

  float sc0[8], sc1[8];
  #pragma unroll
  for (int j = 0; j < 8; ++j) { sc0[j] = 0.0f; sc1[j] = 0.0f; }

  for (int ni = 0; ni < NHALF / 32; ++ni) {
    const int n0 = nhalf * NHALF + ni * 32;
    v8f acc00 = {}, acc01 = {}, acc10 = {}, acc11 = {};
    const __bf16* Bp0 = Binv + (size_t)(n0 + nr) * Dn + hs * 16;  // e = n0+nr
    const __bf16* Bp1 = Bp0 + (size_t)16 * Dn;                    // e = n0+16+nr
    #pragma unroll 2
    for (int k0 = 0; k0 < Dn; k0 += 32) {
      FragBF a0, a1, b0, b1;
      a0.h[0] = *(const v8bf*)(Arow0 + k0);
      a0.h[1] = *(const v8bf*)(Arow0 + k0 + 8);
      a1.h[0] = *(const v8bf*)(Arow1 + k0);
      a1.h[1] = *(const v8bf*)(Arow1 + k0 + 8);
      b0.h[0] = *(const v8bf*)(Bp0 + k0);
      b0.h[1] = *(const v8bf*)(Bp0 + k0 + 8);
      b1.h[0] = *(const v8bf*)(Bp1 + k0);
      b1.h[1] = *(const v8bf*)(Bp1 + k0 + 8);
      acc00 = __builtin_amdgcn_wmma_f32_16x16x32_bf16(
                  false, a0.v, false, b0.v, (short)0, acc00, false, false);
      acc01 = __builtin_amdgcn_wmma_f32_16x16x32_bf16(
                  false, a0.v, false, b1.v, (short)0, acc01, false, false);
      acc10 = __builtin_amdgcn_wmma_f32_16x16x32_bf16(
                  false, a1.v, false, b0.v, (short)0, acc10, false, false);
      acc11 = __builtin_amdgcn_wmma_f32_16x16x32_bf16(
                  false, a1.v, false, b1.v, (short)0, acc11, false, false);
    }
    // fold temp*diff: C layout VGPR j -> M = j + 8*hs, N = nr
    #pragma unroll
    for (int j = 0; j < 8; ++j) {
      int rl0 = rowBase + j + hs * 8;
      int rl1 = rl0 + 16;
      float d00 = (float)A[(size_t)rl0 * Dn + n0 + nr];
      float d01 = (float)A[(size_t)rl0 * Dn + n0 + 16 + nr];
      float d10 = (float)A[(size_t)rl1 * Dn + n0 + nr];
      float d11 = (float)A[(size_t)rl1 * Dn + n0 + 16 + nr];
      sc0[j] += acc00[j] * d00 + acc01[j] * d01;
      sc1[j] += acc10[j] * d10 + acc11[j] * d11;
    }
  }

  // reduce over the 16 lanes of each half-wave, merge N-halves via LDS atomics
  #pragma unroll
  for (int j = 0; j < 8; ++j) {
    float v0 = sc0[j], v1 = sc1[j];
    v0 += __shfl_xor(v0, 1, 16); v1 += __shfl_xor(v1, 1, 16);
    v0 += __shfl_xor(v0, 2, 16); v1 += __shfl_xor(v1, 2, 16);
    v0 += __shfl_xor(v0, 4, 16); v1 += __shfl_xor(v1, 4, 16);
    v0 += __shfl_xor(v0, 8, 16); v1 += __shfl_xor(v1, 8, 16);
    if (nr == 0) {
      atomicAdd(&scorebuf[rowBase + j + hs * 8], v0);        // ds_add_f32
      atomicAdd(&scorebuf[rowBase + 16 + j + hs * 8], v1);
    }
  }
  __syncthreads();

  if (t < BLK_ROWS) {
    int b = rb * BLK_ROWS + t;
    out[(size_t)b * Cn + cls] = -scorebuf[t];
  }
}

extern "C" void kernel_launch(void* const* d_in, const int* in_sizes, int n_in,
                              void* d_out, int out_size, void* d_ws, size_t ws_size,
                              hipStream_t stream) {
  const float* raw  = (const float*)d_in[0];   // [B,D]
  const float* mean = (const float*)d_in[1];   // [C,D]
  const float* inv  = (const float*)d_in[2];   // [C,D,D]
  const float* diag = (const float*)d_in[3];   // [C,D]
  float* out  = (float*)d_out;                 // [B,C]
  __bf16* invT = (__bf16*)d_ws;                // [C,D,D] bf16 transposed

  (void)in_sizes; (void)n_in; (void)out_size; (void)ws_size;

  const int tiles = Cn * (Dn / 32) * (Dn / 32);
  fecam_cvt_transpose<<<tiles, 256, 0, stream>>>(inv, invT);

  const size_t lds = (size_t)BLK_ROWS * Dn * 2 + (size_t)Dn * 4 * 2
                   + (size_t)BLK_ROWS * 4;     // 203264 B
  fecam_score_wmma<<<Cn * (Bn / BLK_ROWS), TPB, lds, stream>>>(
      raw, mean, diag, invT, out);
}
